// ChannelEmbeddingLayers_38740605009948
// MI455X (gfx1250) — matrix operand order repaired and checked by reading, not compiled, not run on hardware
//
#include <hip/hip_runtime.h>
#include <hip/hip_bf16.h>

typedef __attribute__((ext_vector_type(16))) _Float16 v16h;
typedef __attribute__((ext_vector_type(8)))  _Float16 v8h;
typedef __attribute__((ext_vector_type(8)))  float    v8f;

#define BATCH 4096
#define CDIM  2336           // 73 * 32, concat width
#define NOUT  128
#define KT    73             // K tiles of 32
#define MT    (BATCH / 16)   // 256 m-tiles

#define X_BYTES ((size_t)BATCH * CDIM * 2)   // f16 staged concat matrix, ~19.1 MB

// ---------------------------------------------------------------------------
// Kernel 0: repack W[2336][128] f32 -> f16 WMMA B-fragments.
// Fragment layout per (n_tile, k_tile): 32 lanes x 16 halfs contiguous, so the
// GEMM does one fully-coalesced 32B load per lane per fragment.
// B (16-bit, 32x16) lane map: lane = (k_in>>4)*16 + n_in, element j = k_in&15.
// ---------------------------------------------------------------------------
__global__ void prepack_w_kernel(const float* __restrict__ W,
                                 _Float16* __restrict__ Wp) {
    int idx = blockIdx.x * blockDim.x + threadIdx.x;
    if (idx >= CDIM * NOUT) return;
    int k  = idx >> 7;          // 0..2335
    int n  = idx & 127;         // 0..127
    int kt = k >> 5, ki = k & 31;
    int nt = n >> 4, ni = n & 15;
    int lane = ((ki >> 4) << 4) | ni;
    int j    = ki & 15;
    size_t dst = (((size_t)nt * KT + kt) * 32 + lane) * 16 + j;
    Wp[dst] = (_Float16)__builtin_nontemporal_load(&W[idx]);
}

// ---------------------------------------------------------------------------
// Kernel 1: embedding gather + mean-pool + concat -> X[4096][2336] (f16).
// One block per batch row. Table gathers are NON-TEMPORAL: the 256MB/128MB
// tables cannot fit the 192MB L2 and must not evict the staged X/Wp working
// set that the WMMA kernel re-reads from L2.
// Column layout (matches reference concat order):
//   [0,1280)    : dense scalar feats  f=0..19, e=0..63  (col = f*64+e)
//   [1280,1344) : mean over dense seq f=20..69
//   [1344,2304) : sparse scalar feats f=0..29, e=0..31  (col = 1344+f*32+e)
//   [2304,2336) : mean over sparse seq f=30..129
// ---------------------------------------------------------------------------
__global__ void __launch_bounds__(256)
gather_concat_kernel(const int* __restrict__ dkeys,
                     const long long* __restrict__ skeys,
                     const float* __restrict__ dtab,
                     const float* __restrict__ stab,
                     _Float16* __restrict__ X) {
    __shared__ float red[256];
    const int b   = blockIdx.x;
    const int tid = threadIdx.x;
    _Float16* xr = X + (size_t)b * CDIM;
    const int*       dk = dkeys + b * 70;
    const long long* sk = skeys + (size_t)b * 130;

    // dense scalar features (coalesced: a wave covers 32 consecutive e)
    for (int i = tid; i < 1280; i += 256) {
        int f = i >> 6, e = i & 63;
        xr[i] = (_Float16)__builtin_nontemporal_load(
                    &dtab[(size_t)dk[f] * 64 + e]);
    }

    // dense sequence mean: 64 outputs, 50 features, 4 chunks reduced in LDS
    {
        int e = tid & 63, c = tid >> 6;
        float s = 0.f;
        for (int f = 20 + c; f < 70; f += 4)
            s += __builtin_nontemporal_load(&dtab[(size_t)dk[f] * 64 + e]);
        red[tid] = s;
        __syncthreads();
        if (tid < 64) {
            float v = red[tid] + red[tid + 64] + red[tid + 128] + red[tid + 192];
            xr[1280 + tid] = (_Float16)(v * (1.f / 50.f));
        }
        __syncthreads();
    }

    // sparse scalar features
    for (int i = tid; i < 960; i += 256) {
        int f = i >> 5, e = i & 31;
        xr[1344 + i] = (_Float16)__builtin_nontemporal_load(
                           &stab[(size_t)sk[f] * 32 + e]);
    }

    // sparse sequence mean: 32 outputs, 100 features, 8 chunks reduced in LDS
    {
        int e = tid & 31, c = tid >> 5;
        float s = 0.f;
        for (int f = 30 + c; f < 130; f += 8)
            s += __builtin_nontemporal_load(&stab[(size_t)sk[f] * 32 + e]);
        red[tid] = s;
        __syncthreads();
        if (tid < 32) {
            float v = 0.f;
#pragma unroll
            for (int c2 = 0; c2 < 8; ++c2) v += red[tid + 32 * c2];
            xr[2304 + tid] = (_Float16)(v * (1.f / 100.f));
        }
    }
}

// ---------------------------------------------------------------------------
// Kernel 2: Y = relu(X @ W + b) via v_wmma_f32_16x16x32_f16.
// Block = 8 waves; block -> one m-tile (16 rows), wave -> one n-tile (16 cols).
// K loop: 73 WMMA steps. A in contiguous k order (k = 16*hi + s): lane reads two 16B
// chunks of its row; B from prepacked fragments (32B contiguous per lane).
// D layout: VGPR r, lane -> M = (lane>>4)*8 + r, N = lane&15.
// X (19MB) and Wp (598KB) are L2-resident (RT policy loads).
// ---------------------------------------------------------------------------
__global__ void __launch_bounds__(256)
wmma_gemm_kernel(const _Float16* __restrict__ X,
                 const _Float16* __restrict__ Wp,
                 const float* __restrict__ bias,
                 float* __restrict__ out) {
    const int lane = threadIdx.x & 31;
    const int wv   = threadIdx.x >> 5;      // n_tile 0..7
    const int m0   = blockIdx.x << 4;       // m-tile base row
    const int hi   = lane >> 4;
    const int ml   = lane & 15;

    const _Float16* arow  = X  + (size_t)(m0 + ml) * CDIM;
    const _Float16* bfrag = Wp + (size_t)wv * KT * 512 + (size_t)lane * 16;

    v8f acc = {};
    for (int kt = 0; kt < KT; ++kt) {
        const _Float16* ap = arow + kt * 32 + hi * 16;
        v8h alo = *(const v8h*)(ap);
        v8h ahi = *(const v8h*)(ap + 8);
        v16h a  = __builtin_shufflevector(alo, ahi,
                    0, 1, 2, 3, 4, 5, 6, 7, 8, 9, 10, 11, 12, 13, 14, 15);
        v16h bm = *(const v16h*)(bfrag + (size_t)kt * 512);
        acc = __builtin_amdgcn_wmma_f32_16x16x32_f16(
                  /*neg_a=*/false, a, /*neg_b=*/false, bm,
                  /*c_mod=*/(short)0, acc, /*reuse_a=*/false, /*reuse_b=*/false);
    }

    const int   n  = (wv << 4) | ml;
    const float bv = bias[n];
#pragma unroll
    for (int r = 0; r < 8; ++r) {
        float v = acc[r] + bv;
        out[(size_t)(m0 + hi * 8 + r) * NOUT + n] = v > 0.f ? v : 0.f;
    }
}

// ---------------------------------------------------------------------------
extern "C" void kernel_launch(void* const* d_in, const int* in_sizes, int n_in,
                              void* d_out, int out_size, void* d_ws, size_t ws_size,
                              hipStream_t stream) {
    (void)in_sizes; (void)n_in; (void)out_size; (void)ws_size;

    const int*       dkeys = (const int*)d_in[0];        // [4096,70] int32
    const long long* skeys = (const long long*)d_in[1];  // [4096,130] int64
    const float*     dtab  = (const float*)d_in[2];      // [1e6,64] f32
    const float*     stab  = (const float*)d_in[3];      // [1e6,32] f32
    const float*     W     = (const float*)d_in[4];      // [2336,128] f32
    const float*     bias  = (const float*)d_in[5];      // [128] f32
    float*           out   = (float*)d_out;              // [4096,128] f32

    _Float16* X  = (_Float16*)d_ws;                        // 19.1 MB
    _Float16* Wp = (_Float16*)((char*)d_ws + X_BYTES);     // 598 KB fragments

    prepack_w_kernel<<<(CDIM * NOUT + 255) / 256, 256, 0, stream>>>(W, Wp);
    gather_concat_kernel<<<BATCH, 256, 0, stream>>>(dkeys, skeys, dtab, stab, X);
    wmma_gemm_kernel<<<MT, 256, 0, stream>>>(X, Wp, bias, out);
}
